// MultiheadedAttention_68195490726480
// MI455X (gfx1250) — compile-verified, matrix operand order checked by reading, not run on hardware
//
#include <hip/hip_runtime.h>

// Problem constants (reference: T,B,E,H = 2048,2,1024,16; S = 64)
#define T_DIM 2048
#define B_DIM 2
#define E_DIM 1024
#define H_DIM 16
#define S_DIM 64

typedef __attribute__((ext_vector_type(16))) __bf16 bf16x16;
typedef __attribute__((ext_vector_type(8)))  float  f32x8;
typedef __attribute__((ext_vector_type(4)))  int    i32x4;

union FragU {
    bf16x16 bf;
    uint4   q[2];
};

// ---------------------------------------------------------------------------
// gfx1250 async global->LDS path (GLOBAL_LOAD_ASYNC_TO_LDS_B128, ASYNCcnt).
// Builtin signature (from compiler diagnostic): arg0 = int4 AS(1)*,
// arg1 = int4 AS(3)*, then imm offset, imm cpol.
// Guarded so the build falls back to plain copies if the builtins are absent.
// ---------------------------------------------------------------------------
#if defined(__HIP_DEVICE_COMPILE__) && defined(__gfx1250__) && \
    __has_builtin(__builtin_amdgcn_global_load_async_to_lds_b128) && \
    __has_builtin(__builtin_amdgcn_s_wait_asynccnt)
#define USE_ASYNC_LDS 1
#else
#define USE_ASYNC_LDS 0
#endif

typedef __attribute__((address_space(1))) i32x4 gas_i32x4;
typedef __attribute__((address_space(3))) i32x4 las_i32x4;

__device__ __forceinline__ gas_i32x4* to_gas16(const void* p) {
    // generic global pointer bits == global address; inttoptr into AS1
    return (gas_i32x4*)(unsigned long long)(uintptr_t)p;
}
__device__ __forceinline__ las_i32x4* to_las16(void* p) {
    // generic LDS pointer: low 32 bits are the LDS byte offset (aperture rules)
    return (las_i32x4*)(unsigned int)(uintptr_t)p;
}

__device__ __forceinline__ void copy16_g2l(const unsigned short* g, unsigned short* l) {
#if USE_ASYNC_LDS
    __builtin_amdgcn_global_load_async_to_lds_b128(to_gas16(g), to_las16(l), 0, 0);
#else
    *(uint4*)l = *(const uint4*)g;
#endif
}
__device__ __forceinline__ void async_wait0() {
#if USE_ASYNC_LDS
    __builtin_amdgcn_s_wait_asynccnt(0);
#endif
}

__device__ __forceinline__ unsigned short f2bf(float f) {
    unsigned int u = __float_as_uint(f);
    u += 0x7FFFu + ((u >> 16) & 1u);   // round-to-nearest-even
    return (unsigned short)(u >> 16);
}

// A/B fragment loader for V_WMMA_*_16X16X32_BF16.
// Per ISA 16-bit A layout (CDNA5 ISA): lane L holds row (L&15);
// lanes 0-15 hold K = kk*32 + {0..7, 16..23}, lanes 16-31 K = kk*32 + {8..15, 24..31}.
// For B, a lane's column n = (L&15) maps to row n of a row-major operand with
// contiguous K — identical access pattern. Two 16-byte loads per fragment.
__device__ __forceinline__ bf16x16 load_frag_row16(const unsigned short* base, int ld, int kk) {
    int lane = threadIdx.x & 31;
    int row  = lane & 15;
    int off  = (lane < 16) ? 0 : 8;
    const unsigned short* p = base + row * ld + kk * 32 + off;
    FragU f;
    f.q[0] = *(const uint4*)(p);
    f.q[1] = *(const uint4*)(p + 16);
    return f.bf;
}

#define WMMA_BF16(acc, a, b) \
    acc = __builtin_amdgcn_wmma_f32_16x16x32_bf16(false, (a), false, (b), (short)0, (acc), false, false)

__device__ __forceinline__ float hmax16(float v) {
    v = fmaxf(v, __shfl_xor(v, 1, 32));
    v = fmaxf(v, __shfl_xor(v, 2, 32));
    v = fmaxf(v, __shfl_xor(v, 4, 32));
    v = fmaxf(v, __shfl_xor(v, 8, 32));
    return v;
}
__device__ __forceinline__ float hsum16(float v) {
    v += __shfl_xor(v, 1, 32);
    v += __shfl_xor(v, 2, 32);
    v += __shfl_xor(v, 4, 32);
    v += __shfl_xor(v, 8, 32);
    return v;
}

// ---------------------------------------------------------------------------
// Kernel 1: Q/K'/V projection.  K'[r,d] = K[r,d]/32 + Er[d,r]  (bias folding).
// Grid: B*H*(T/64) = 1024 blocks, 256 threads (8 waves).
// Wave w owns row strip mi = w>>1 and column tiles ni = (2w)&3, (2w+1)&3 —
// the matrix loop (Q/K/V) is fully unrolled so there is no divergent `mat`.
// ---------------------------------------------------------------------------
__global__ __launch_bounds__(256) void qkv_kernel(
    const float* __restrict__ x,  const float* __restrict__ Wq,
    const float* __restrict__ Wk, const float* __restrict__ Wv,
    const float* __restrict__ Er,
    unsigned short* __restrict__ qb, unsigned short* __restrict__ kb,
    unsigned short* __restrict__ vb) {
    __shared__ __attribute__((aligned(16))) unsigned short Xs[64 * 72];
    __shared__ __attribute__((aligned(16))) unsigned short Ws[3][64 * 72];

    int blk = blockIdx.x;
    int tt  = blk & 31;            // T/64 = 32
    int h   = (blk >> 5) & 15;
    int b   = blk >> 9;
    int t0  = tt * 64;
    int tid = threadIdx.x;

    // Stage X tile [64 t x 64 s] (bf16), coalesced along s.
    for (int i = tid; i < 4096; i += 256) {
        int m = i >> 6, s = i & 63;
        Xs[m * 72 + s] =
            f2bf(x[(size_t)(t0 + m) * (B_DIM * E_DIM) + b * E_DIM + h * 64 + s]);
    }
    {
        const float* Wsrc[3] = {Wq + h * 4096, Wk + h * 4096, Wv + h * 4096};
        #pragma unroll
        for (int w3 = 0; w3 < 3; ++w3)
            for (int i = tid; i < 4096; i += 256) {
                int d = i >> 6, s = i & 63;
                Ws[w3][d * 72 + s] = f2bf(Wsrc[w3][d * 64 + s]);
            }
    }
    __syncthreads();

    int wave  = tid >> 5, lane = tid & 31;
    int colL  = lane & 15;
    int rbase = (lane < 16) ? 0 : 8;   // 16x16 f32 C/D layout: lanes 16-31 hold rows M+8
    size_t base = (size_t)(b * H_DIM + h) * T_DIM * 64;

    int mi = wave >> 1;                       // same row strip for both tiles
    bf16x16 a0 = load_frag_row16(Xs + mi * 16 * 72, 72, 0);
    bf16x16 a1 = load_frag_row16(Xs + mi * 16 * 72, 72, 1);
    int tg0 = t0 + mi * 16 + rbase;           // first of 8 consecutive output rows

    #pragma unroll
    for (int u = 0; u < 2; ++u) {
        int ni  = (wave * 2 + u) & 3;
        int col = ni * 16 + colL;

        f32x8 accq = {}, acck = {}, accv = {};
        WMMA_BF16(accq, a0, load_frag_row16(Ws[0] + ni * 16 * 72, 72, 0));
        WMMA_BF16(accq, a1, load_frag_row16(Ws[0] + ni * 16 * 72, 72, 1));
        WMMA_BF16(acck, a0, load_frag_row16(Ws[1] + ni * 16 * 72, 72, 0));
        WMMA_BF16(acck, a1, load_frag_row16(Ws[1] + ni * 16 * 72, 72, 1));
        WMMA_BF16(accv, a0, load_frag_row16(Ws[2] + ni * 16 * 72, 72, 0));
        WMMA_BF16(accv, a1, load_frag_row16(Ws[2] + ni * 16 * 72, 72, 1));

        // Er[d, t]: 8 consecutive t values -> two float4 loads (32B aligned).
        const float* ep = Er + (size_t)col * T_DIM + tg0;
        float4 e0 = *(const float4*)(ep);
        float4 e1 = *(const float4*)(ep + 4);
        float ev[8] = {e0.x, e0.y, e0.z, e0.w, e1.x, e1.y, e1.z, e1.w};

        #pragma unroll
        for (int r = 0; r < 8; ++r) {
            size_t o = base + (size_t)(tg0 + r) * 64 + col;
            qb[o] = f2bf(accq[r]);
            kb[o] = f2bf(acck[r] * (1.0f / 32.0f) + ev[r]);
            vb[o] = f2bf(accv[r]);
        }
    }
}

// ---------------------------------------------------------------------------
// Kernel 2: causal flash attention.  scores = Q @ K'^T (bias already folded).
// Grid: B*H*(T/64) = 1024 blocks, 128 threads (4 waves; each owns 16 q-rows).
// K' tile is staged with GLOBAL_LOAD_ASYNC_TO_LDS_B128 and overlaps with the
// VALU transpose-staging of V^T.
// ---------------------------------------------------------------------------
__global__ __launch_bounds__(128) void attn_kernel(
    const unsigned short* __restrict__ qb, const unsigned short* __restrict__ kb,
    const unsigned short* __restrict__ vb, unsigned short* __restrict__ ab) {
    __shared__ __attribute__((aligned(16))) unsigned short Ks[64 * 72];      // K' row-major [r][d]
    __shared__ __attribute__((aligned(16))) unsigned short Vts[64 * 72];     // V transposed [d][r]
    __shared__ __attribute__((aligned(16))) unsigned short Ps[4][16 * 72];   // per-wave P repack

    int blk = blockIdx.x;
    int tt  = blk & 31;
    int h   = (blk >> 5) & 15;
    int b   = blk >> 9;
    int t0  = tt * 64;
    int tid = threadIdx.x;
    int wave = tid >> 5, lane = tid & 31;
    int colL = lane & 15;
    int rbase = (lane < 16) ? 0 : 8;
    size_t hb = (size_t)(b * H_DIM + h) * T_DIM * 64;

    // Preload this wave's Q strip fragments (16 x 64) straight from global bf16.
    bf16x16 qf0, qf1;
    {
        const unsigned short* qrow = qb + hb + (size_t)(t0 + wave * 16) * 64;
        qf0 = load_frag_row16(qrow, 64, 0);
        qf1 = load_frag_row16(qrow, 64, 1);
    }

    f32x8 O0 = {}, O1 = {}, O2 = {}, O3 = {};
    f32x8 mrun, lrun;
    #pragma unroll
    for (int r = 0; r < 8; ++r) { mrun[r] = -1e30f; lrun[r] = 0.0f; }

    for (int r0 = 0; r0 <= t0; r0 += 64) {            // causal: only blocks <= t0
        __syncthreads();
        // Async-stage K' (64 rows x 128B = 512 x 16B chunks, 4 per thread)...
        for (int i = tid; i < 512; i += 128) {
            int j = i >> 3, c = i & 7;
            copy16_g2l(kb + hb + (size_t)(r0 + j) * 64 + c * 8, Ks + j * 72 + c * 8);
        }
        // ...overlapped with the VALU transpose-stage of V^T.
        for (int i = tid; i < 4096; i += 128) {
            int j = i >> 6, d = i & 63;
            Vts[d * 72 + j] = vb[hb + (size_t)(r0 + j) * 64 + d];
        }
        async_wait0();
        __syncthreads();

        // S = Q @ K'^T : four 16x16 tiles along keys.
        f32x8 S[4];
        #pragma unroll
        for (int n = 0; n < 4; ++n) {
            f32x8 acc = {};
            WMMA_BF16(acc, qf0, load_frag_row16(Ks + n * 16 * 72, 72, 0));
            WMMA_BF16(acc, qf1, load_frag_row16(Ks + n * 16 * 72, 72, 1));
            S[n] = acc;
        }

        // Causal mask — only the diagonal block can violate col <= row.
        if (r0 == t0) {
            #pragma unroll
            for (int n = 0; n < 4; ++n) {
                int col = r0 + n * 16 + colL;
                #pragma unroll
                for (int r = 0; r < 8; ++r) {
                    int tg = t0 + wave * 16 + r + rbase;
                    if (col > tg) S[n][r] = -1e9f;
                }
            }
        }

        // Online softmax: stats live in v8f aligned with the C-layout rows.
        f32x8 mnew, corr;
        #pragma unroll
        for (int r = 0; r < 8; ++r) {
            float mx = fmaxf(fmaxf(S[0][r], S[1][r]), fmaxf(S[2][r], S[3][r]));
            mx = hmax16(mx);
            mnew[r] = fmaxf(mrun[r], mx);
            corr[r] = __expf(mrun[r] - mnew[r]);
        }
        #pragma unroll
        for (int n = 0; n < 4; ++n)
            #pragma unroll
            for (int r = 0; r < 8; ++r)
                S[n][r] = __expf(S[n][r] - mnew[r]);
        #pragma unroll
        for (int r = 0; r < 8; ++r) {
            float sm = S[0][r] + S[1][r] + S[2][r] + S[3][r];
            sm = hsum16(sm);
            lrun[r] = lrun[r] * corr[r] + sm;
            mrun[r] = mnew[r];
            O0[r] *= corr[r]; O1[r] *= corr[r]; O2[r] *= corr[r]; O3[r] *= corr[r];
        }

        // Repack P: C-layout (row=vgpr, col=lane) -> A-layout via wave-private LDS.
        unsigned short* Pw = Ps[wave];
        #pragma unroll
        for (int n = 0; n < 4; ++n)
            #pragma unroll
            for (int r = 0; r < 8; ++r)
                Pw[(r + rbase) * 72 + n * 16 + colL] = f2bf(S[n][r]);

        // O += P @ V  (B fragments from V^T rows => contiguous loads).
        #pragma unroll
        for (int kk = 0; kk < 2; ++kk) {
            bf16x16 a = load_frag_row16(Pw, 72, kk);
            WMMA_BF16(O0, a, load_frag_row16(Vts + 0 * 16 * 72, 72, kk));
            WMMA_BF16(O1, a, load_frag_row16(Vts + 1 * 16 * 72, 72, kk));
            WMMA_BF16(O2, a, load_frag_row16(Vts + 2 * 16 * 72, 72, kk));
            WMMA_BF16(O3, a, load_frag_row16(Vts + 3 * 16 * 72, 72, kk));
        }
    }

    // Normalize and store in the quirk layout: ab[b][e = h*64+d][t]  (bf16).
    #pragma unroll
    for (int r = 0; r < 8; ++r) {
        float inv = 1.0f / lrun[r];
        O0[r] *= inv; O1[r] *= inv; O2[r] *= inv; O3[r] *= inv;
    }
    size_t abase = (size_t)b * E_DIM * T_DIM + (size_t)(h * 64) * T_DIM;
    f32x8* Os[4] = {&O0, &O1, &O2, &O3};
    #pragma unroll
    for (int n = 0; n < 4; ++n) {
        int d = n * 16 + colL;
        #pragma unroll
        for (int r = 0; r < 8; ++r) {
            int tg = t0 + wave * 16 + r + rbase;
            ab[abase + (size_t)d * T_DIM + tg] = f2bf((*Os[n])[r]);
        }
    }
}

// ---------------------------------------------------------------------------
// Kernel 3: output projection  Y = X @ Wr^T + br, where X is the ab buffer
// read *linearly* as [T][E] per batch (this IS the reference's reshape quirk).
// Grid: B*(T/64)*(E/64) = 1024 blocks, 256 threads (8 waves, 2 tiles each).
// X tile staged async (already bf16); Wr tile converted fp32->bf16 on VALU.
// ---------------------------------------------------------------------------
__global__ __launch_bounds__(256) void proj_kernel(
    const unsigned short* __restrict__ ab, const float* __restrict__ Wr,
    const float* __restrict__ br, float* __restrict__ out) {
    __shared__ __attribute__((aligned(16))) unsigned short Xs[64 * 72];
    __shared__ __attribute__((aligned(16))) unsigned short Ws[64 * 72];

    int blk = blockIdx.x;
    int dd = blk & 15;            // E/64 = 16
    int it = (blk >> 4) & 31;     // T/64 = 32
    int b  = blk >> 9;
    int d0 = dd * 64, i0 = it * 64;
    int tid = threadIdx.x;
    int wave = tid >> 5, lane = tid & 31;
    int colL = lane & 15;
    int rbase = (lane < 16) ? 0 : 8;
    int mi = wave & 3;            // row strip owned by this wave
    int nbase = (wave >> 2) * 2;  // two 16-col tiles owned by this wave

    f32x8 acc0 = {}, acc1 = {};
    const unsigned short* xb = ab + (size_t)b * E_DIM * T_DIM;

    for (int jc = 0; jc < 16; ++jc) {        // K = 1024 = 16 x 64
        int j0 = jc * 64;
        __syncthreads();
        // Async-stage X (512 x 16B chunks, 2 per thread)...
        for (int i = tid; i < 512; i += 256) {
            int m = i >> 3, c = i & 7;
            copy16_g2l(xb + (size_t)(i0 + m) * E_DIM + j0 + c * 8, Xs + m * 72 + c * 8);
        }
        // ...overlapped with fp32->bf16 conversion of the Wr tile.
        for (int i = tid; i < 4096; i += 256) {
            int m = i >> 6, k = i & 63;
            Ws[m * 72 + k] = f2bf(Wr[(size_t)(d0 + m) * E_DIM + j0 + k]);
        }
        async_wait0();
        __syncthreads();
        #pragma unroll
        for (int kk = 0; kk < 2; ++kk) {
            bf16x16 a = load_frag_row16(Xs + mi * 16 * 72, 72, kk);
            WMMA_BF16(acc0, a, load_frag_row16(Ws + (nbase + 0) * 16 * 72, 72, kk));
            WMMA_BF16(acc1, a, load_frag_row16(Ws + (nbase + 1) * 16 * 72, 72, kk));
        }
    }

    int d_0 = d0 + (nbase + 0) * 16 + colL;
    int d_1 = d0 + (nbase + 1) * 16 + colL;
    float bias0 = br[d_0], bias1 = br[d_1];
    size_t obase = (size_t)b * T_DIM * E_DIM;
    #pragma unroll
    for (int r = 0; r < 8; ++r) {
        int ig = i0 + mi * 16 + r + rbase;
        out[obase + (size_t)ig * E_DIM + d_0] = acc0[r] + bias0;
        out[obase + (size_t)ig * E_DIM + d_1] = acc1[r] + bias1;
    }
}

// ---------------------------------------------------------------------------
extern "C" void kernel_launch(void* const* d_in, const int* in_sizes, int n_in,
                              void* d_out, int out_size, void* d_ws, size_t ws_size,
                              hipStream_t stream) {
    const float* x  = (const float*)d_in[0];
    const float* Wq = (const float*)d_in[1];
    const float* Wk = (const float*)d_in[2];
    const float* Wv = (const float*)d_in[3];
    const float* Er = (const float*)d_in[4];
    const float* Wr = (const float*)d_in[5];
    const float* br = (const float*)d_in[6];
    float* out = (float*)d_out;

    // Workspace: Q, K', V as [B][H][T][S] bf16; attn out as [B][E][T] bf16.
    const size_t n = (size_t)B_DIM * H_DIM * T_DIM * S_DIM;   // 4Mi elems each
    unsigned short* qb = (unsigned short*)d_ws;
    unsigned short* kb = qb + n;
    unsigned short* vb = kb + n;
    unsigned short* ab = vb + n;
    (void)in_sizes; (void)n_in; (void)out_size; (void)ws_size;

    qkv_kernel <<<B_DIM * H_DIM * (T_DIM / 64), 256, 0, stream>>>(x, Wq, Wk, Wv, Er, qb, kb, vb);
    attn_kernel<<<B_DIM * H_DIM * (T_DIM / 64), 128, 0, stream>>>(qb, kb, vb, ab);
    proj_kernel<<<B_DIM * (T_DIM / 64) * (E_DIM / 64), 256, 0, stream>>>(ab, Wr, br, out);
}
